// RecursiveGraphConvolutionStepAddOn_90726889161219
// MI455X (gfx1250) — matrix operand order repaired and checked by reading, not compiled, not run on hardware
//
#include <hip/hip_runtime.h>
#include <hip/hip_bf16.h>

// ---- problem constants ----
constexpr int NN  = 10000;   // nodes
constexpr int DD  = 512;     // feature dim
constexpr int HH  = 256;     // hidden dim
constexpr int OO  = 128;     // out dim
constexpr int NO2 = 256;     // 2*OO (concat mean|log_std)
constexpr int KT_ADJ = (NN + 31) / 32;   // 313 (remainder tile of 16)
constexpr int RBLK = 32;                 // rows per workgroup
constexpr int NBLKS = (NN + RBLK - 1) / RBLK;  // 313
constexpr int SSTR = 40;                 // LDS row stride (bf16 elems), padded

// ---- raw (trivial) vector types for WMMA + wide ld/st ----
typedef __attribute__((ext_vector_type(16))) __bf16 v16bf;
typedef __attribute__((ext_vector_type(8)))  float  v8f;
typedef __attribute__((ext_vector_type(4)))  float  f32x4;
typedef __attribute__((ext_vector_type(4)))  unsigned int u32x4;

union BF16x16 { v16bf v; u32x4 q[2]; };
union BF16x8  { u32x4 q;  __bf16 h[8]; };
union BF16x4  { unsigned long long q; __bf16 h[4]; };

__device__ __forceinline__ v16bf load_b16x16(const __bf16* p) {
  BF16x16 t;
  const u32x4* q = (const u32x4*)p;
  t.q[0] = q[0];
  t.q[1] = q[1];
  return t.v;
}

__device__ __forceinline__ v16bf zero16() {
  BF16x16 z;
  z.q[0] = (u32x4){0, 0, 0, 0};
  z.q[1] = (u32x4){0, 0, 0, 0};
  return z.v;
}

__device__ __forceinline__ v8f wmma_bf16(v16bf a, v16bf b, v8f c) {
  return __builtin_amdgcn_wmma_f32_16x16x32_bf16(
      false, a, false, b, (short)0, c, false, false);
}

__device__ __forceinline__ void put4(float* d, f32x4 v) {
  d[0] = v[0]; d[1] = v[1]; d[2] = v[2]; d[3] = v[3];
}

// ---------------------------------------------------------------------------
// Kernel 1: dinv[i] = rsqrt(1 + sum_j adj[i][j])
// ---------------------------------------------------------------------------
__global__ void k_dinv(const float* __restrict__ adj, float* __restrict__ dinv) {
  const int row = blockIdx.x;
  const f32x4* r4 = (const f32x4*)(adj + (size_t)row * NN);
  float s = 0.0f;
  for (int i = threadIdx.x; i < NN / 4; i += blockDim.x) {
    f32x4 v = r4[i];
    s += v[0] + v[1] + v[2] + v[3];
  }
  __shared__ float red[256];
  red[threadIdx.x] = s;
  __syncthreads();
  for (int off = 128; off > 0; off >>= 1) {
    if ((int)threadIdx.x < off) red[threadIdx.x] += red[threadIdx.x + off];
    __syncthreads();
  }
  if (threadIdx.x == 0) dinv[row] = rsqrtf(red[0] + 1.0f);
}

// ---------------------------------------------------------------------------
// Kernel 2: weight prep (bf16, column-major = B-operand friendly)
// ---------------------------------------------------------------------------
__global__ void k_prep(const float* __restrict__ W1, const float* __restrict__ Wm,
                       const float* __restrict__ Ws,
                       __bf16* __restrict__ W1T, __bf16* __restrict__ WcT) {
  const int id = blockIdx.x * blockDim.x + threadIdx.x;
  if (id < HH * DD) {
    const int n = id / DD, k = id % DD;
    W1T[id] = (__bf16)W1[(size_t)k * HH + n];
  }
  if (id < NO2 * HH) {
    const int n = id / HH, k = id % HH;
    const float v = (n < OO) ? Wm[(size_t)k * OO + n] : Ws[(size_t)k * OO + (n - OO)];
    WcT[id] = (__bf16)v;
  }
}

// ---------------------------------------------------------------------------
// Kernel 3: xwT = (X @ W1)^T, bf16 col-major [256][10000]
// 8 waves per block: 2 row-tiles x 4 col-groups; 4 accum tiles per wave.
// ---------------------------------------------------------------------------
__global__ void k_xw(const float* __restrict__ x, const __bf16* __restrict__ W1T,
                     __bf16* __restrict__ xwT) {
  const int lane = threadIdx.x & 31;
  const int wave = threadIdx.x >> 5;
  const int mL = lane & 15, hi = lane >> 4;
  const int rt = wave & 1, cg = wave >> 1;
  const int rowW = blockIdx.x * RBLK + rt * 16;
  const int rowA = (rowW + mL < NN) ? (rowW + mL) : (NN - 1);
  const float* arow = x + (size_t)rowA * DD;

  v8f acc[4] = {};
  for (int kt = 0; kt < DD / 32; ++kt) {
    const int kb = kt * 32;
    const int k0 = kb + hi * 16;
    float sa[16];
    put4(sa + 0,  *(const f32x4*)(arow + k0));
    put4(sa + 4,  *(const f32x4*)(arow + k0 + 4));
    put4(sa + 8,  *(const f32x4*)(arow + k0 + 8));
    put4(sa + 12, *(const f32x4*)(arow + k0 + 12));
    BF16x16 A;
#pragma unroll
    for (int e = 0; e < 16; ++e) A.v[e] = (__bf16)sa[e];
#pragma unroll
    for (int j = 0; j < 4; ++j) {
      const int n = cg * 64 + j * 16 + mL;
      v16bf B = load_b16x16(W1T + (size_t)n * DD + kb + hi * 16);
      acc[j] = wmma_bf16(A.v, B, acc[j]);
    }
  }
  if (rowW < NN) {
#pragma unroll
    for (int j = 0; j < 4; ++j) {
      const int n = cg * 64 + j * 16 + mL;
      BF16x8 o;
#pragma unroll
      for (int r = 0; r < 8; ++r) o.h[r] = (__bf16)acc[j][r];
      *(u32x4*)(xwT + (size_t)n * NN + rowW + hi * 8) = o.q;
    }
  }
}

// ---------------------------------------------------------------------------
// Kernels 4/6: AN @ B  (B = xwT -> relu -> h bf16 row-major;
//                       B = gT  -> f32 split z_mean|z_log_std)
// adj block (32x32) staged once per workgroup in LDS (bf16, pre-scaled),
// double-buffered; each wave computes 16 rows x 64 cols (4 WMMA tiles).
// ---------------------------------------------------------------------------
template <bool IS_H>
__global__ void k_an(const float* __restrict__ adj, const float* __restrict__ dinv,
                     const __bf16* __restrict__ Bm, __bf16* __restrict__ hOut,
                     float* __restrict__ zOut) {
  __shared__ alignas(16) __bf16 sA[2][RBLK * SSTR];

  const int t = threadIdx.x;
  const int lane = t & 31;
  const int wave = t >> 5;
  const int mL = lane & 15, hi = lane >> 4;
  const int rt = wave & 1, cg = wave >> 1;
  const int row0 = blockIdx.x * RBLK;
  const int rowW = row0 + rt * 16;

  // staging role: 32 rows x 8 chunks of 4 cols
  const int sRow = t >> 3;
  const int sC4 = (t & 7) * 4;
  const int gRow = row0 + sRow;
  const bool rowOk = (gRow < NN);
  const float drow = rowOk ? dinv[gRow] : 0.0f;
  const float* aRowPtr = adj + (size_t)(rowOk ? gRow : 0) * NN;

  v8f acc[4] = {};

  // stage kt = 0
  {
    f32x4 a = {0.f, 0.f, 0.f, 0.f}, d = {0.f, 0.f, 0.f, 0.f};
    if (rowOk) {
      a = *(const f32x4*)(aRowPtr + sC4);
      d = *(const f32x4*)(dinv + sC4);
    }
    BF16x4 p;
#pragma unroll
    for (int e = 0; e < 4; ++e) p.h[e] = (__bf16)(a[e] * d[e] * drow);
    *(unsigned long long*)&sA[0][sRow * SSTR + sC4] = p.q;
  }
  __syncthreads();

  for (int kt = 0; kt < KT_ADJ; ++kt) {
    const int kb = kt * 32;

    // stage kt+1 into the other buffer
    if (kt + 1 < KT_ADJ) {
      const int gc = kb + 32 + sC4;
      __builtin_prefetch(aRowPtr + gc + 256, 0, 1);   // stream adj ~8 iters ahead
      f32x4 a = {0.f, 0.f, 0.f, 0.f}, d = {0.f, 0.f, 0.f, 0.f};
      if (rowOk && gc < NN) {
        a = *(const f32x4*)(aRowPtr + gc);
        d = *(const f32x4*)(dinv + gc);
      }
      BF16x4 p;
#pragma unroll
      for (int e = 0; e < 4; ++e) p.h[e] = (__bf16)(a[e] * d[e] * drow);
      *(unsigned long long*)&sA[(kt + 1) & 1][sRow * SSTR + sC4] = p.q;
    }

    // A fragment from LDS (two ds_load_b128 per lane)
    const char* ap = (const char*)&sA[kt & 1][(rt * 16 + mL) * SSTR];
    BF16x16 A;
    A.q[0] = *(const u32x4*)(ap + hi * 32);        // k = kb + hi*16 + 0..7
    A.q[1] = *(const u32x4*)(ap + 16 + hi * 32);   // k = kb + hi*16 + 8..15

#pragma unroll
    for (int j = 0; j < 4; ++j) {
      const int n = cg * 64 + j * 16 + mL;
      v16bf B;
      if (kb + hi * 16 < NN) B = load_b16x16(Bm + (size_t)n * NN + kb + hi * 16);
      else                   B = zero16();
      acc[j] = wmma_bf16(A.v, B, acc[j]);
    }
    __syncthreads();
  }

  // epilogue: + dinv_i^2 * B[i][n]  (diag of AN), then activation/store
  const int rBase = rowW + hi * 8;
  const int rClamp = (rBase + 8 <= NN) ? rBase : (NN - 8);  // safe read base
  float dd[8];
  put4(dd + 0, *(const f32x4*)(dinv + rClamp));
  put4(dd + 4, *(const f32x4*)(dinv + rClamp + 4));
#pragma unroll
  for (int j = 0; j < 4; ++j) {
    const int n = cg * 64 + j * 16 + mL;
    BF16x8 diag;
    diag.q = *(const u32x4*)(Bm + (size_t)n * NN + rClamp);
    if (rBase < NN) {
#pragma unroll
      for (int r = 0; r < 8; ++r) {
        float v = acc[j][r] + dd[r] * dd[r] * (float)diag.h[r];
        const int row = rBase + r;
        if (IS_H) {
          v = fmaxf(v, 0.0f);
          hOut[(size_t)row * NO2 + n] = (__bf16)v;
        } else {
          if (n < OO) zOut[(size_t)row * OO + n] = v;
          else        zOut[(size_t)NN * OO + (size_t)row * OO + (n - OO)] = v;
        }
      }
    }
  }
}

// ---------------------------------------------------------------------------
// Kernel 5: gT = (H @ [Wm|Ws])^T, bf16 col-major [256][10000]
// ---------------------------------------------------------------------------
__global__ void k_g(const __bf16* __restrict__ hIn, const __bf16* __restrict__ WcT,
                    __bf16* __restrict__ gT) {
  const int lane = threadIdx.x & 31;
  const int wave = threadIdx.x >> 5;
  const int mL = lane & 15, hi = lane >> 4;
  const int rt = wave & 1, cg = wave >> 1;
  const int rowW = blockIdx.x * RBLK + rt * 16;
  const int rowA = (rowW + mL < NN) ? (rowW + mL) : (NN - 1);
  const __bf16* arow = hIn + (size_t)rowA * HH;

  v8f acc[4] = {};
  for (int kt = 0; kt < HH / 32; ++kt) {
    const int kb = kt * 32;
    const int k0 = kb + hi * 16;
    BF16x8 p0, p1;
    p0.q = *(const u32x4*)(arow + k0);
    p1.q = *(const u32x4*)(arow + k0 + 8);
    BF16x16 A;
#pragma unroll
    for (int e = 0; e < 8; ++e) { A.v[e] = p0.h[e]; A.v[8 + e] = p1.h[e]; }
#pragma unroll
    for (int j = 0; j < 4; ++j) {
      const int n = cg * 64 + j * 16 + mL;
      v16bf B = load_b16x16(WcT + (size_t)n * HH + kb + hi * 16);
      acc[j] = wmma_bf16(A.v, B, acc[j]);
    }
  }
  if (rowW < NN) {
#pragma unroll
    for (int j = 0; j < 4; ++j) {
      const int n = cg * 64 + j * 16 + mL;
      BF16x8 o;
#pragma unroll
      for (int r = 0; r < 8; ++r) o.h[r] = (__bf16)acc[j][r];
      *(u32x4*)(gT + (size_t)n * NN + rowW + hi * 8) = o.q;
    }
  }
}

// ---------------------------------------------------------------------------
extern "C" void kernel_launch(void* const* d_in, const int* in_sizes, int n_in,
                              void* d_out, int out_size, void* d_ws, size_t ws_size,
                              hipStream_t stream) {
  const float* adj = (const float*)d_in[0];
  const float* x   = (const float*)d_in[1];
  const float* W1  = (const float*)d_in[2];
  const float* Wm  = (const float*)d_in[3];
  const float* Ws  = (const float*)d_in[4];
  float* out = (float*)d_out;

  // workspace layout (bytes), ~15.8 MB total
  char* ws = (char*)d_ws;
  float*  dinv = (float*)(ws + 0);                    //    40,000
  __bf16* W1T  = (__bf16*)(ws + 40960);               //   262,144
  __bf16* WcT  = (__bf16*)(ws + 40960 + 262144);      //   131,072
  __bf16* xwT  = (__bf16*)(ws + 434176);              // 5,120,000
  __bf16* hM   = (__bf16*)(ws + 5554176);             // 5,120,000
  __bf16* gT   = (__bf16*)(ws + 10674176);            // 5,120,000

  k_dinv<<<NN, 256, 0, stream>>>(adj, dinv);
  k_prep<<<(HH * DD + 255) / 256, 256, 0, stream>>>(W1, Wm, Ws, W1T, WcT);

  k_xw<<<NBLKS, 256, 0, stream>>>(x, W1T, xwT);
  k_an<true><<<NBLKS, 256, 0, stream>>>(adj, dinv, xwT, hM, nullptr);
  k_g<<<NBLKS, 256, 0, stream>>>(hM, WcT, gT);
  k_an<false><<<NBLKS, 256, 0, stream>>>(adj, dinv, gT, nullptr, out);
}
